// BiMambaEncoder_56495999811862
// MI455X (gfx1250) — matrix operand order repaired and checked by reading, not compiled, not run on hardware
//
#include <hip/hip_runtime.h>
#include <hip/hip_bf16.h>

// ---------------- model dims ----------------
constexpr int BSZ     = 2;
constexpr int SEQ     = 512;
constexpr int DMODEL  = 768;
constexpr int DSTATE  = 64;
constexpr int HEADDIM = 64;
constexpr int DINNER  = 1536;            // 2*DMODEL
constexpr int NHEADS  = 24;              // DINNER/HEADDIM
constexpr int CONVDIM = DINNER + 2 * DSTATE;          // 1664
constexpr int DINPROJ = 2 * DINNER + 2 * DSTATE + NHEADS; // 3224
constexpr int DFF     = 4 * DMODEL;      // 3072
constexpr int MTOK    = BSZ * SEQ;       // 1024 rows for all GEMMs
constexpr float EPSV  = 1e-5f;

typedef __attribute__((ext_vector_type(16))) _Float16 v16h;
typedef __attribute__((ext_vector_type(8)))  _Float16 v8h;
typedef __attribute__((ext_vector_type(8)))  float    v8f;

// ---------------- helpers ----------------
__device__ __forceinline__ float block_reduce_sum256(float v, float* s) {
  #pragma unroll
  for (int o = 16; o; o >>= 1) v += __shfl_xor(v, o, 32);   // wave32 reduce
  int lane = threadIdx.x & 31, wv = threadIdx.x >> 5;
  __syncthreads();
  if (lane == 0) s[wv] = v;
  __syncthreads();
  float r = 0.f;
  #pragma unroll
  for (int i = 0; i < 8; ++i) r += s[i];
  return r;
}

// A fragment: 16x32 f16, row-major source [M,K]. Lane l (l<16): M=row0+l,
// K = k0 + {0..7,16..23}; lanes 16-31: same rows, K pattern +8. (ISA 7.12.2)
__device__ __forceinline__ v16h load_frag_a(const _Float16* __restrict__ A, int ld,
                                            int row0, int k0, int maxRow) {
  int lane = threadIdx.x & 31;
  int r = row0 + (lane & 15);
  r = (r < maxRow) ? r : (maxRow - 1);           // select, not branch
  int kb = k0 + ((lane >> 4) << 3);
  const _Float16* p = A + (size_t)r * ld + kb;
  v8h lo = *(const v8h*)(p);
  v8h hi = *(const v8h*)(p + 16);
  v16h f;
  #pragma unroll
  for (int e = 0; e < 8; ++e) { f[e] = lo[e]; f[8 + e] = hi[e]; }
  return f;
}

// B fragment: 32x16 f16 (KxN) from weight W[N,K] row-major. Lane = N column,
// K = k0 + {0..7,16..23}; lanes 16-31: K pattern +8 (matches A) -> two b128 loads.
__device__ __forceinline__ v16h load_frag_b(const _Float16* __restrict__ W, int ld,
                                            int n0, int k0, int maxRow) {
  int lane = threadIdx.x & 31;
  int r = n0 + (lane & 15);
  r = (r < maxRow) ? r : (maxRow - 1);
  int kb = k0 + ((lane >> 4) << 3);
  const _Float16* p = W + (size_t)r * ld + kb;
  v8h lo = *(const v8h*)(p);
  v8h hi = *(const v8h*)(p + 16);
  v16h f;
  #pragma unroll
  for (int e = 0; e < 8; ++e) { f[e] = lo[e]; f[8 + e] = hi[e]; }
  return f;
}

// ---------------- kernels ----------------
__global__ void cast_f16_kernel(const float* __restrict__ in, _Float16* __restrict__ out, int n) {
  for (int i = blockIdx.x * blockDim.x + threadIdx.x; i < n; i += gridDim.x * blockDim.x)
    out[i] = (_Float16)in[i];
}

// LayerNorm over 768, one block (256 thr) per row, f16 output for WMMA feed.
__global__ __launch_bounds__(256)
void layernorm768_kernel(const float* __restrict__ x, const float* __restrict__ w,
                         const float* __restrict__ b, _Float16* __restrict__ out16) {
  __shared__ float sred[8];
  const float* row = x + (size_t)blockIdx.x * DMODEL;
  float v[3], s = 0.f, s2 = 0.f;
  #pragma unroll
  for (int i = 0; i < 3; ++i) {
    int c = threadIdx.x + i * 256;
    v[i] = row[c]; s += v[i]; s2 += v[i] * v[i];
  }
  s  = block_reduce_sum256(s, sred);
  s2 = block_reduce_sum256(s2, sred);
  float mu  = s * (1.f / DMODEL);
  float var = s2 * (1.f / DMODEL) - mu * mu;
  float inv = rsqrtf(var + EPSV);
  _Float16* o = out16 + (size_t)blockIdx.x * DMODEL;
  #pragma unroll
  for (int i = 0; i < 3; ++i) {
    int c = threadIdx.x + i * 256;
    o[c] = (_Float16)((v[i] - mu) * inv * w[c] + b[c]);
  }
}

// C[M,N] (f32) = act( A_f16[M,K] @ W_f16[N,K]^T + bias ) [+ addsrc] [+ C]
// 8 waves/block: wave tile 32x32 (2x2 WMMA frags), block tile 64x128.
__global__ __launch_bounds__(256)
void gemm_wmma_kernel(const _Float16* __restrict__ A, const _Float16* __restrict__ W,
                      const float* __restrict__ bias, const float* __restrict__ addsrc,
                      float* __restrict__ C, int M, int N, int K,
                      int act, int accumulate) {
  int w  = threadIdx.x >> 5;
  int wm = w & 1, wn = w >> 1;
  int mBase = blockIdx.y * 64  + wm * 32;
  int nBase = blockIdx.x * 128 + wn * 32;

  v8f acc00 = {}, acc01 = {}, acc10 = {}, acc11 = {};

  for (int k0 = 0; k0 < K; k0 += 32) {
    if (k0 + 32 < K) {  // uniform condition: L2 prefetch of next K slab
      __builtin_prefetch(A + (size_t)mBase * K + k0 + 32, 0, 1);
      __builtin_prefetch(W + (size_t)nBase * K + k0 + 32, 0, 1);
    }
    v16h a0 = load_frag_a(A, K, mBase,      k0, M);
    v16h a1 = load_frag_a(A, K, mBase + 16, k0, M);
    v16h b0 = load_frag_b(W, K, nBase,      k0, N);
    v16h b1 = load_frag_b(W, K, nBase + 16, k0, N);
    acc00 = __builtin_amdgcn_wmma_f32_16x16x32_f16(false, a0, false, b0, (short)0, acc00, false, false);
    acc01 = __builtin_amdgcn_wmma_f32_16x16x32_f16(false, a0, false, b1, (short)0, acc01, false, false);
    acc10 = __builtin_amdgcn_wmma_f32_16x16x32_f16(false, a1, false, b0, (short)0, acc10, false, false);
    acc11 = __builtin_amdgcn_wmma_f32_16x16x32_f16(false, a1, false, b1, (short)0, acc11, false, false);
  }

  // C/D layout: lane = N col (mod 16), VGPR r -> M = r + 8*(lane>=16)
  int lane   = threadIdx.x & 31;
  int rowOff = (lane >> 4) * 8;
  int colLo  = lane & 15;
  v8f* accs[2][2] = { { &acc00, &acc01 }, { &acc10, &acc11 } };
  #pragma unroll
  for (int i = 0; i < 2; ++i) {
    #pragma unroll
    for (int j = 0; j < 2; ++j) {
      int col = nBase + j * 16 + colLo;
      if (col < N) {
        float bi = bias ? bias[col] : 0.f;
        #pragma unroll
        for (int r = 0; r < 8; ++r) {
          int row = mBase + i * 16 + rowOff + r;
          if (row < M) {
            float v = (*accs[i][j])[r] + bi;
            if (act == 1) v = 0.5f * v * (1.f + erff(v * 0.70710678118f));  // exact GELU
            size_t idx = (size_t)row * N + col;
            if (addsrc)     v += addsrc[idx];
            if (accumulate) v += C[idx];
            C[idx] = v;
          }
        }
      }
    }
  }
}

// Causal depthwise conv(4) + SiLU on xBC channels; softplus(dt)+dA precompute.
// grid = (MTOK, 2 dirs); backward dir reads zxbcdt rows flipped in L.
__global__ __launch_bounds__(256)
void conv_silu_dt_kernel(const float* __restrict__ zxbcdt,
                         const float* __restrict__ conv_w,   // [CONVDIM,4]
                         const float* __restrict__ conv_b,   // [CONVDIM]
                         const float* __restrict__ dt_bias,  // [NHEADS]
                         const float* __restrict__ A_log,    // [NHEADS]
                         float* __restrict__ xact,           // [2][B][L][CONVDIM]
                         float* __restrict__ dtpost,         // [2][B][L][NHEADS]
                         float* __restrict__ dAbuf) {        // [2][B][L][NHEADS]
  int dir = blockIdx.y;
  int b = blockIdx.x / SEQ, l = blockIdx.x % SEQ;
  auto src_row = [&](int ll) -> const float* {
    int ls = dir ? (SEQ - 1 - ll) : ll;
    return zxbcdt + ((size_t)b * SEQ + ls) * DINPROJ;
  };
  float* orow = xact + (((size_t)dir * BSZ + b) * SEQ + l) * CONVDIM;
  for (int c = threadIdx.x; c < CONVDIM; c += 256) {
    float acc = conv_b[c];
    #pragma unroll
    for (int k = 0; k < 4; ++k) {
      int j = l - 3 + k;                          // causal: pad (3,0)
      if (j >= 0) acc += src_row(j)[DINNER + c] * conv_w[c * 4 + k];
    }
    orow[c] = acc / (1.f + expf(-acc));           // SiLU
  }
  if (threadIdx.x < NHEADS) {
    int h = threadIdx.x;
    float dtraw = src_row(l)[DINNER + CONVDIM + h] + dt_bias[h];
    float sp = (dtraw > 20.f) ? dtraw : log1pf(expf(dtraw));  // softplus
    size_t o = (((size_t)dir * BSZ + b) * SEQ + l) * NHEADS + h;
    dtpost[o] = sp;
    dAbuf[o]  = expf(-expf(A_log[h]) * sp);
  }
}

// Sequential selective scan. 96 blocks = 2dir x 2batch x 24heads, 256 thr.
// Thread (p=t>>2, q=t&3) owns 16 state elems h[p][16q..16q+15] in registers.
__global__ __launch_bounds__(256)
void ssm_scan_kernel(const float* __restrict__ xact, const float* __restrict__ dtpost,
                     const float* __restrict__ dAbuf, const float* __restrict__ D_param,
                     float* __restrict__ ydir) {
  int dir = blockIdx.x & 1;
  int bh  = blockIdx.x >> 1;
  int b = bh / NHEADS, h = bh % NHEADS;
  int p = threadIdx.x >> 2, q = threadIdx.x & 3, n0 = q * 16;

  float s[16];
  #pragma unroll
  for (int j = 0; j < 16; ++j) s[j] = 0.f;
  float Dp = D_param[h];

  const float* base = xact  + (((size_t)dir * BSZ + b) * SEQ) * CONVDIM;
  const float* dtb  = dtpost + (((size_t)dir * BSZ + b) * SEQ) * NHEADS;
  const float* dab  = dAbuf  + (((size_t)dir * BSZ + b) * SEQ) * NHEADS;
  float*       yo   = ydir   + (((size_t)dir * BSZ + b) * SEQ) * DINNER;

  for (int l = 0; l < SEQ; ++l) {
    const float* row = base + (size_t)l * CONVDIM;
    float xp = row[h * HEADDIM + p];
    const float4* Bp = (const float4*)(row + DINNER + n0);
    const float4* Cp = (const float4*)(row + DINNER + DSTATE + n0);
    float Bv[16], Cv[16];
    #pragma unroll
    for (int t = 0; t < 4; ++t) {
      float4 f = Bp[t]; Bv[4*t] = f.x; Bv[4*t+1] = f.y; Bv[4*t+2] = f.z; Bv[4*t+3] = f.w;
      float4 g = Cp[t]; Cv[4*t] = g.x; Cv[4*t+1] = g.y; Cv[4*t+2] = g.z; Cv[4*t+3] = g.w;
    }
    float dtv = dtb[(size_t)l * NHEADS + h];
    float dav = dab[(size_t)l * NHEADS + h];
    float coef = dtv * xp;
    float accv = 0.f;
    #pragma unroll
    for (int j = 0; j < 16; ++j) {
      s[j] = dav * s[j] + coef * Bv[j];   // h = dA*h + dt*B*x
      accv += s[j] * Cv[j];               // y partial = h . C
    }
    accv += __shfl_xor(accv, 1, 32);      // 4-lane reduce within wave32
    accv += __shfl_xor(accv, 2, 32);
    if (q == 0) yo[(size_t)l * DINNER + h * HEADDIM + p] = accv + Dp * xp;
  }
}

// y * silu(z) -> RMSNorm * norm_w -> f16, written at ORIGINAL seq position
// so both directions feed out_proj aligned. grid = (MTOK, 2).
__global__ __launch_bounds__(256)
void gated_rmsnorm_kernel(const float* __restrict__ ydir, const float* __restrict__ zxbcdt,
                          const float* __restrict__ norm_w, _Float16* __restrict__ yln) {
  __shared__ float sred[8];
  int dir = blockIdx.y;
  int b = blockIdx.x / SEQ, l = blockIdx.x % SEQ;
  int ls = dir ? (SEQ - 1 - l) : l;     // original position of dir-local l
  const float* y = ydir   + (((size_t)dir * BSZ + b) * SEQ + l) * DINNER;
  const float* z = zxbcdt + ((size_t)b * SEQ + ls) * DINPROJ;   // z = first DINNER
  float g[6], ss = 0.f;
  #pragma unroll
  for (int i = 0; i < 6; ++i) {
    int c = threadIdx.x + i * 256;
    float zv = z[c];
    float gv = y[c] * (zv / (1.f + expf(-zv)));
    g[i] = gv; ss += gv * gv;
  }
  ss = block_reduce_sum256(ss, sred);
  float inv = rsqrtf(ss * (1.f / DINNER) + EPSV);
  _Float16* o = yln + (((size_t)dir * BSZ + b) * SEQ + ls) * DINNER;
  #pragma unroll
  for (int i = 0; i < 6; ++i) {
    int c = threadIdx.x + i * 256;
    o[c] = (_Float16)(g[i] * inv * norm_w[c]);
  }
}

// ---------------- host orchestration ----------------
extern "C" void kernel_launch(void* const* d_in, const int* in_sizes, int n_in,
                              void* d_out, int out_size, void* d_ws, size_t ws_size,
                              hipStream_t stream) {
  const float* x          = (const float*)d_in[0];
  const float* in_proj_w  = (const float*)d_in[1];
  const float* conv_w     = (const float*)d_in[2];
  const float* conv_b     = (const float*)d_in[3];
  const float* dt_bias    = (const float*)d_in[4];
  const float* A_log      = (const float*)d_in[5];
  const float* D_param    = (const float*)d_in[6];
  const float* norm_w     = (const float*)d_in[7];
  const float* out_proj_w = (const float*)d_in[8];
  const float* ln1_w      = (const float*)d_in[9];
  const float* ln1_b      = (const float*)d_in[10];
  const float* ln2_w      = (const float*)d_in[11];
  const float* ln2_b      = (const float*)d_in[12];
  const float* ff_w1      = (const float*)d_in[13];
  const float* ff_b1      = (const float*)d_in[14];
  const float* ff_w2      = (const float*)d_in[15];
  const float* ff_b2      = (const float*)d_in[16];
  float* out = (float*)d_out;

  char* wsp = (char*)d_ws;
  size_t off = 0;
  auto alloc = [&](size_t bytes) -> void* {
    void* p = wsp + off;
    off += (bytes + 255) & ~(size_t)255;
    return p;
  };

  _Float16* xn16     = (_Float16*)alloc((size_t)MTOK * DMODEL * 2);
  _Float16* w_in16   = (_Float16*)alloc((size_t)DINPROJ * DMODEL * 2);
  _Float16* w_out16  = (_Float16*)alloc((size_t)DMODEL * DINNER * 2);
  _Float16* w_ff1_16 = (_Float16*)alloc((size_t)DFF * DMODEL * 2);
  _Float16* w_ff2_16 = (_Float16*)alloc((size_t)DMODEL * DFF * 2);
  float*    zxbcdt   = (float*)alloc((size_t)MTOK * DINPROJ * 4);
  float*    xactb    = (float*)alloc((size_t)2 * MTOK * CONVDIM * 4);
  float*    dtpost   = (float*)alloc((size_t)2 * MTOK * NHEADS * 4);
  float*    dAbuf    = (float*)alloc((size_t)2 * MTOK * NHEADS * 4);
  float*    ydir     = (float*)alloc((size_t)2 * MTOK * DINNER * 4);
  _Float16* yln16    = (_Float16*)alloc((size_t)2 * MTOK * DINNER * 2);
  float*    mambao   = (float*)alloc((size_t)MTOK * DMODEL * 4);
  _Float16* m16      = (_Float16*)alloc((size_t)MTOK * DMODEL * 2);
  float*    ff1      = (float*)alloc((size_t)MTOK * DFF * 4);
  _Float16* ff1_16   = (_Float16*)alloc((size_t)MTOK * DFF * 2);

  // weight casts (L2-resident f16 copies)
  cast_f16_kernel<<<256, 256, 0, stream>>>(in_proj_w,  w_in16,   DINPROJ * DMODEL);
  cast_f16_kernel<<<256, 256, 0, stream>>>(out_proj_w, w_out16,  DMODEL * DINNER);
  cast_f16_kernel<<<256, 256, 0, stream>>>(ff_w1,      w_ff1_16, DFF * DMODEL);
  cast_f16_kernel<<<256, 256, 0, stream>>>(ff_w2,      w_ff2_16, DMODEL * DFF);

  // LN1 -> f16
  layernorm768_kernel<<<MTOK, 256, 0, stream>>>(x, ln1_w, ln1_b, xn16);

  // in_proj GEMM (computed once; backward dir is just a row flip of this)
  dim3 gIn((DINPROJ + 127) / 128, MTOK / 64);
  gemm_wmma_kernel<<<gIn, 256, 0, stream>>>(xn16, w_in16, nullptr, nullptr,
                                            zxbcdt, MTOK, DINPROJ, DMODEL, 0, 0);

  // per-direction conv+silu + dt/dA precompute
  conv_silu_dt_kernel<<<dim3(MTOK, 2), 256, 0, stream>>>(zxbcdt, conv_w, conv_b,
                                                         dt_bias, A_log,
                                                         xactb, dtpost, dAbuf);
  // selective scan
  ssm_scan_kernel<<<2 * BSZ * NHEADS, 256, 0, stream>>>(xactb, dtpost, dAbuf, D_param, ydir);

  // gate + RMSNorm -> f16 (original order)
  gated_rmsnorm_kernel<<<dim3(MTOK, 2), 256, 0, stream>>>(ydir, zxbcdt, norm_w, yln16);

  // out_proj: forward writes, backward accumulates
  dim3 gOut((DMODEL + 127) / 128, MTOK / 64);
  gemm_wmma_kernel<<<gOut, 256, 0, stream>>>(yln16, w_out16, nullptr, nullptr,
                                             mambao, MTOK, DMODEL, DINNER, 0, 0);
  gemm_wmma_kernel<<<gOut, 256, 0, stream>>>(yln16 + (size_t)MTOK * DINNER, w_out16,
                                             nullptr, nullptr,
                                             mambao, MTOK, DMODEL, DINNER, 0, 1);

  // LN2 -> f16
  layernorm768_kernel<<<MTOK, 256, 0, stream>>>(mambao, ln2_w, ln2_b, m16);

  // FFN
  dim3 gF1((DFF + 127) / 128, MTOK / 64);
  gemm_wmma_kernel<<<gF1, 256, 0, stream>>>(m16, w_ff1_16, ff_b1, nullptr,
                                            ff1, MTOK, DFF, DMODEL, /*gelu*/1, 0);
  cast_f16_kernel<<<256, 256, 0, stream>>>(ff1, ff1_16, MTOK * DFF);
  gemm_wmma_kernel<<<gOut, 256, 0, stream>>>(ff1_16, w_ff2_16, ff_b2, /*residual*/x,
                                             out, MTOK, DMODEL, DFF, 0, 0);
}
